// RSSM_46943992545351
// MI455X (gfx1250) — compile-verified, matrix operand order checked by reading, not run on hardware
//
#include <hip/hip_runtime.h>
#include <hip/hip_bf16.h>

typedef __bf16 bf16;
typedef __attribute__((ext_vector_type(16))) __bf16 v16bf;
typedef __attribute__((ext_vector_type(8)))  float  v8f;
typedef int v4i __attribute__((ext_vector_type(4)));

#define LN_EPS 1e-3f

#define AS1 __attribute__((address_space(1)))
#define AS3 __attribute__((address_space(3)))

#if defined(__has_builtin)
# if __has_builtin(__builtin_amdgcn_global_load_async_to_lds_b128)
#  define USE_ASYNC_LDS 1
# endif
# if __has_builtin(__builtin_amdgcn_s_wait_asynccnt)
#  define HAVE_WAIT_ASYNC_BUILTIN 1
# endif
#endif

__device__ __forceinline__ void wait_async0() {
#if defined(USE_ASYNC_LDS)
# if defined(HAVE_WAIT_ASYNC_BUILTIN)
    __builtin_amdgcn_s_wait_asynccnt(0);
# else
    asm volatile("s_wait_asynccnt 0x0" ::: "memory");
# endif
#endif
}

// ---------------------------------------------------------------------------
// Fragment loading per CDNA5 ISA 16-bit A/B layout (CDNA5 ISA):
// lanes 0-15 hold K = koff+[0..7] and koff+[16..23] of row (lane&15);
// lanes 16-31 hold K = koff+[8..15] and koff+[24..31].
// B stored transposed [N][K] so the same loader serves A and B.
// ---------------------------------------------------------------------------
union Frag16 { uint4 u4[2]; v16bf v; };

__device__ __forceinline__ v16bf load_frag(const bf16* __restrict__ base, int strideElems,
                                           int koff, int lane) {
    const int half = (lane >> 4) & 1;
    const int r    = lane & 15;
    const bf16* p  = base + (size_t)r * strideElems + koff + half * 8;
    Frag16 f;
    f.u4[0] = *reinterpret_cast<const uint4*>(p);        // K = koff + half*8 + [0..7]
    f.u4[1] = *reinterpret_cast<const uint4*>(p + 16);   // K = koff + 16 + half*8 + [0..7]
    return f.v;
}

// Same K-pattern, but source is f32 in LDS; convert to bf16 in registers.
__device__ __forceinline__ v16bf load_frag_f32(const float* __restrict__ base, int strideElems,
                                               int koff, int lane) {
    const int half = (lane >> 4) & 1;
    const int r    = lane & 15;
    const float* p = base + (size_t)r * strideElems + koff + half * 8;
    float4 f0 = *reinterpret_cast<const float4*>(p);
    float4 f1 = *reinterpret_cast<const float4*>(p + 4);
    float4 f2 = *reinterpret_cast<const float4*>(p + 16);
    float4 f3 = *reinterpret_cast<const float4*>(p + 20);
    v16bf v;
    v[0] = (bf16)f0.x; v[1] = (bf16)f0.y; v[2]  = (bf16)f0.z; v[3]  = (bf16)f0.w;
    v[4] = (bf16)f1.x; v[5] = (bf16)f1.y; v[6]  = (bf16)f1.z; v[7]  = (bf16)f1.w;
    v[8] = (bf16)f2.x; v[9] = (bf16)f2.y; v[10] = (bf16)f2.z; v[11] = (bf16)f2.w;
    v[12]= (bf16)f3.x; v[13]= (bf16)f3.y; v[14] = (bf16)f3.z; v[15] = (bf16)f3.w;
    return v;
}

__device__ __forceinline__ float wred_sum(float x) {
    #pragma unroll
    for (int o = 16; o; o >>= 1) x += __shfl_xor(x, o, 32);
    return x;
}
__device__ __forceinline__ float wred_max(float x) {
    #pragma unroll
    for (int o = 16; o; o >>= 1) x = fmaxf(x, __shfl_xor(x, o, 32));
    return x;
}
__device__ __forceinline__ float sigmoidf(float x) { return 1.f / (1.f + expf(-x)); }
__device__ __forceinline__ float siluf(float x)    { return x / (1.f + expf(-x)); }

// ---------------------------------------------------------------------------
// 16-row GEMM, K-outer / N-tiles-in-registers:
// wave w owns N-tiles {w, w+16, ..., w+16*(TPW-1)}; all TPW accumulators stay
// live in VGPRs. Per K-chunk: 1 A-fragment (ds_load) feeds TPW WMMAs; TPW
// B-fragments stream from L2 with immediate-offset folds.  No value is
// loop-invariant across any loop, so nothing gets hoisted/spilled.
// A = [A1 | A2]; A2 may be f32 in LDS (converted in registers).
// ---------------------------------------------------------------------------
template<int TPW, bool A2F32>
__device__ __forceinline__ void gemm_rows16(
    const bf16* __restrict__ a1, int a1Stride, int K1,
    const void* __restrict__ a2v, int a2Stride,
    const bf16* __restrict__ wt, int Ktot,
    float* __restrict__ outAcc, int outStride,
    int wave, int lane)
{
    const bf16*  a2b = (const bf16*)a2v;
    const float* a2f = (const float*)a2v;
    const bf16* bbase = wt + (size_t)wave * 16 * Ktot;   // tile j at +j*16*16*Ktot elems

    v8f acc[TPW];
    #pragma unroll
    for (int j = 0; j < TPW; ++j)
        acc[j] = (v8f){0.f,0.f,0.f,0.f,0.f,0.f,0.f,0.f};

    #pragma unroll 2
    for (int kw = 0; kw < Ktot; kw += 32) {
        v16bf a = (kw < K1) ? load_frag(a1, a1Stride, kw, lane)
                : (A2F32    ? load_frag_f32(a2f, a2Stride, kw - K1, lane)
                            : load_frag(a2b, a2Stride, kw - K1, lane));
        v16bf b[TPW];
        #pragma unroll
        for (int j = 0; j < TPW; ++j)
            b[j] = load_frag(bbase + (size_t)j * 256 * Ktot, Ktot, kw, lane);
        #pragma unroll
        for (int j = 0; j < TPW; ++j)
            acc[j] = __builtin_amdgcn_wmma_f32_16x16x32_bf16(false, a, false, b[j],
                                                             (short)0, acc[j], false, false);
    }

    // C/D layout: VGPR r, lanes 0-15 -> M=r, N=lane; lanes 16-31 -> M=8+r, N=lane-16
    const int n  = lane & 15;
    const int mb = (lane >> 4) * 8;
    #pragma unroll
    for (int j = 0; j < TPW; ++j) {
        float* dst = outAcc + (wave + j * 16) * 16 + n;
        #pragma unroll
        for (int r = 0; r < 8; ++r)
            dst[(size_t)(mb + r) * outStride] = acc[j][r];
    }
}

// Wave-per-row LayerNorm (N=512) + SiLU, f32 src -> bf16 dst
__device__ __forceinline__ void ln_silu_row(const float* __restrict__ src,
                                            const float* __restrict__ g,
                                            const float* __restrict__ b,
                                            bf16* __restrict__ dst, int lane)
{
    float s = 0.f, s2 = 0.f;
    #pragma unroll
    for (int i = lane; i < 512; i += 32) { float v = src[i]; s += v; s2 += v * v; }
    s = wred_sum(s); s2 = wred_sum(s2);
    const float mean = s * (1.f / 512.f);
    const float inv  = rsqrtf(s2 * (1.f / 512.f) - mean * mean + LN_EPS);
    #pragma unroll
    for (int i = lane; i < 512; i += 32) {
        float v = (src[i] - mean) * inv * g[i] + b[i];
        dst[i] = (bf16)siluf(v);
    }
}

// Wave-per-row unimix softmax head: 32 groups of D=32 (one lane per d).
__device__ __forceinline__ void stat_head(const float* __restrict__ accRow,
                                          const float* __restrict__ bias,
                                          float* __restrict__ outLogit,
                                          float* __restrict__ outStoch,
                                          unsigned char* idxRow, int lane)
{
    for (int sg = 0; sg < 32; ++sg) {
        const int c = sg * 32 + lane;
        const float l = accRow[c] + bias[c];
        const float m = wred_max(l);
        const float e = expf(l - m);
        const float se = wred_sum(e);
        float p = e / se;
        p = 0.99f * p + (0.01f / 32.f);
        unsigned long long bal = __ballot(l == m);
        const int arg = __ffsll(bal) - 1;           // first max = jnp.argmax semantics
        outLogit[c] = logf(p);
        outStoch[c] = (lane == arg) ? 1.f : 0.f;
        if (idxRow && lane == 0) idxRow[sg] = (unsigned char)arg;
    }
}

// ---------------------------------------------------------------------------
// Prep kernel 1: transpose + convert f32 [K][N] -> bf16 [N][K]
// ---------------------------------------------------------------------------
__global__ void transpose_bf16(const float* __restrict__ src, bf16* __restrict__ dst,
                               int K, int N) {
    const int id = blockIdx.x * blockDim.x + threadIdx.x;
    if (id >= K * N) return;
    const int n = id / K, k = id % K;
    dst[(size_t)n * K + k] = (bf16)src[(size_t)k * N + n];
}

// ---------------------------------------------------------------------------
// Prep kernel 2: deter0 = tanh(W_init); h0 = silu(ln(deter0 @ W_img_out));
// idx0 = per-group argmax of logit0 (stoch0 is exactly one-hot of idx0).
// ---------------------------------------------------------------------------
__global__ __launch_bounds__(512) void init_state(
    const float* __restrict__ W_init, const float* __restrict__ W_img_out,
    const float* __restrict__ g_io, const float* __restrict__ b_io,
    const float* __restrict__ W_ims, const float* __restrict__ b_ims,
    float* __restrict__ deter0, int* __restrict__ idx0)
{
    __shared__ float sh_d[512];
    __shared__ float sh_pre[512];
    __shared__ float sh_h[512];
    __shared__ float sh_l[1024];
    const int tid = threadIdx.x;
    float d0 = tanhf(W_init[tid]);
    deter0[tid] = d0;
    sh_d[tid] = d0;
    __syncthreads();
    float acc = 0.f;
    for (int k = 0; k < 512; ++k) acc += sh_d[k] * W_img_out[k * 512 + tid];
    sh_pre[tid] = acc;
    __syncthreads();
    float s = 0.f, s2 = 0.f;
    for (int k = 0; k < 512; ++k) { float v = sh_pre[k]; s += v; s2 += v * v; }
    const float mean = s * (1.f / 512.f);
    const float inv  = rsqrtf(s2 * (1.f / 512.f) - mean * mean + LN_EPS);
    sh_h[tid] = siluf((acc - mean) * inv * g_io[tid] + b_io[tid]);
    __syncthreads();
    for (int c = tid; c < 1024; c += 512) {
        float a = b_ims[c];
        for (int k = 0; k < 512; ++k) a += sh_h[k] * W_ims[k * 1024 + c];
        sh_l[c] = a;
    }
    __syncthreads();
    if (tid < 32) {
        float best = sh_l[tid * 32]; int arg = 0;
        for (int d = 1; d < 32; ++d) {
            float x = sh_l[tid * 32 + d];
            if (x > best) { best = x; arg = d; }
        }
        idx0[tid] = arg;
    }
}

// ---------------------------------------------------------------------------
// Main kernel: 32 blocks x 16 batch rows; each block runs the full T=64 scan
// with no inter-block sync (all stages are row-local in batch).
// ---------------------------------------------------------------------------
__global__ __launch_bounds__(512) void rssm_main(
    const float* __restrict__ embed, const float* __restrict__ action,
    const float* __restrict__ is_first, const float* __restrict__ W_img_in,
    const float* __restrict__ g_img_in, const float* __restrict__ b_img_in,
    const float* __restrict__ g_gru,    const float* __restrict__ b_gru,
    const float* __restrict__ g_img_out,const float* __restrict__ b_img_out,
    const float* __restrict__ g_obs_out,const float* __restrict__ b_obs_out,
    const float* __restrict__ b_ims_stat,const float* __restrict__ b_obs_stat,
    const bf16* __restrict__ wt_gru,     const bf16* __restrict__ wt_img_out,
    const bf16* __restrict__ wt_ims_stat,const bf16* __restrict__ wt_obs_out,
    const bf16* __restrict__ wt_obs_stat,
    const float* __restrict__ deter0, const int* __restrict__ idx0,
    float* __restrict__ out)
{
    __shared__ __align__(16) bf16  s_A2[16 * 1024];    // [x | deter_bf16], A for GEMM2/3/5
    __shared__ __align__(16) float s_embf[16 * 1024];  // f32 embedding staging (async target)
    __shared__ __align__(16) bf16  s_v512[16 * 512];   // h (GEMM4 A) then x2 (GEMM6 A)
    __shared__ __align__(16) float s_deter[16 * 512];  // f32 recurrent state
    __shared__ __align__(16) float s_acc[16 * 1536];   // f32 GEMM output scratch
    __shared__ float s_act[16 * 8];
    __shared__ unsigned char s_idx[16 * 32];           // one-hot stoch as indices

    const int tid  = threadIdx.x;
    const int wave = tid >> 5;
    const int lane = tid & 31;
    const int w    = wave;                  // row owned in row-parallel phases
    const int b0   = blockIdx.x * 16;

    // initialize recurrent state (is_first[:,0]==1 would do this anyway; be safe)
    for (int i = lane; i < 512; i += 32) {
        float d0 = deter0[i];
        s_deter[w * 512 + i] = d0;
        s_A2[w * 1024 + 512 + i] = (bf16)d0;
    }
    s_idx[w * 32 + lane] = (unsigned char)idx0[lane];
    __syncthreads();

    for (int t = 0; t < 64; ++t) {
        const size_t bt   = (size_t)(b0 + w) * 64 + t;
        float* __restrict__ outp = out + bt * 4608;

        // ---- preamble: kick off async embed copy; is_first reset; masked action
#if defined(USE_ASYNC_LDS)
        {
            const float* gsrc = embed + bt * 1024;          // this wave's batch row
            #pragma unroll
            for (int i = lane * 4; i < 1024; i += 128) {    // 16B per lane per issue
                __builtin_amdgcn_global_load_async_to_lds_b128(
                    (AS1 v4i*)(gsrc + i),
                    (AS3 v4i*)(s_embf + w * 1024 + i), 0, 0);
            }
        }
#else
        for (int i = lane; i < 1024; i += 32)
            s_embf[w * 1024 + i] = embed[bt * 1024 + i];
#endif
        const bool first = (is_first[bt] != 0.f);
        if (lane < 8) s_act[w * 8 + lane] =
            (lane < 7 && !first) ? action[bt * 7 + lane] : 0.f;
        if (first) {
            s_idx[w * 32 + lane] = (unsigned char)idx0[lane];
            for (int i = lane; i < 512; i += 32) {
                float d0 = deter0[i];
                s_deter[w * 512 + i] = d0;
                s_A2[w * 1024 + 512 + i] = (bf16)d0;
            }
        }
        __syncthreads();

        // ---- stage 1: x = silu(ln([onehot-stoch | act] @ W_img_in))
        // stoch is exactly one-hot -> gather 32 weight rows + K=7 action part.
        {
            float xp[16];
            #pragma unroll
            for (int j = 0; j < 16; ++j) xp[j] = 0.f;
            for (int sg = 0; sg < 32; ++sg) {
                const float* wrow =
                    W_img_in + (size_t)(sg * 32 + s_idx[w * 32 + sg]) * 512;
                #pragma unroll
                for (int j = 0; j < 16; ++j) xp[j] += wrow[lane + 32 * j];
            }
            #pragma unroll
            for (int a = 0; a < 7; ++a) {
                const float av = s_act[w * 8 + a];
                const float* wrow = W_img_in + (size_t)(1024 + a) * 512;
                #pragma unroll
                for (int j = 0; j < 16; ++j) xp[j] += av * wrow[lane + 32 * j];
            }
            float s = 0.f, s2 = 0.f;
            #pragma unroll
            for (int j = 0; j < 16; ++j) { s += xp[j]; s2 += xp[j] * xp[j]; }
            s = wred_sum(s); s2 = wred_sum(s2);
            const float mean = s * (1.f / 512.f);
            const float inv  = rsqrtf(s2 * (1.f / 512.f) - mean * mean + LN_EPS);
            #pragma unroll
            for (int j = 0; j < 16; ++j) {
                const int c = lane + 32 * j;
                float v = (xp[j] - mean) * inv * g_img_in[c] + b_img_in[c];
                s_A2[w * 1024 + c] = (bf16)siluf(v);
            }
        }
        __syncthreads();

        // ---- GEMM2: parts = [x | deter] @ W_gru   (K=1024, N=1536)
        gemm_rows16<6, false>(s_A2, 1024, 1024, nullptr, 0,
                              wt_gru, 1024, s_acc, 1536, wave, lane);
        __syncthreads();

        // ---- GRU cell: LN over 1536, split, update deter (f32 state)
        {
            const float* p = s_acc + w * 1536;
            float s = 0.f, s2 = 0.f;
            for (int i = lane; i < 1536; i += 32) { float v = p[i]; s += v; s2 += v * v; }
            s = wred_sum(s); s2 = wred_sum(s2);
            const float mean = s * (1.f / 1536.f);
            const float inv  = rsqrtf(s2 * (1.f / 1536.f) - mean * mean + LN_EPS);
            for (int i = lane; i < 512; i += 32) {
                float r = (p[i]        - mean) * inv * g_gru[i]        + b_gru[i];
                float c = (p[512 + i]  - mean) * inv * g_gru[512 + i]  + b_gru[512 + i];
                float u = (p[1024 + i] - mean) * inv * g_gru[1024 + i] + b_gru[1024 + i];
                r = sigmoidf(r);
                c = tanhf(r * c);
                u = sigmoidf(u - 1.f);                      // UPDATE_BIAS
                const float dn = u * c + (1.f - u) * s_deter[w * 512 + i];
                s_deter[w * 512 + i] = dn;
                s_A2[w * 1024 + 512 + i] = (bf16)dn;
                outp[4096 + i] = dn;                        // ys deter slot
            }
        }
        __syncthreads();

        // ---- GEMM3: deter_n @ W_img_out (K=512, N=512)
        gemm_rows16<2, false>(s_A2 + 512, 1024, 512, nullptr, 0,
                              wt_img_out, 512, s_acc, 512, wave, lane);
        __syncthreads();
        ln_silu_row(s_acc + w * 512, g_img_out, b_img_out, s_v512 + w * 512, lane);
        __syncthreads();

        // ---- GEMM4: h @ W_ims_stat (K=512, N=1024) -> prior head
        gemm_rows16<4, false>(s_v512, 512, 512, nullptr, 0,
                              wt_ims_stat, 512, s_acc, 1024, wave, lane);
        __syncthreads();
        stat_head(s_acc + w * 1024, b_ims_stat, outp + 2048, outp + 3072, nullptr, lane);
        wait_async0();                 // embed staging complete before GEMM5 reads it
        __syncthreads();

        // ---- GEMM5: [deter_n | emb] @ W_obs_out (K=1536, N=512); emb read f32->bf16
        gemm_rows16<2, true>(s_A2 + 512, 1024, 512, s_embf, 1024,
                             wt_obs_out, 1536, s_acc, 512, wave, lane);
        __syncthreads();
        ln_silu_row(s_acc + w * 512, g_obs_out, b_obs_out, s_v512 + w * 512, lane);
        __syncthreads();

        // ---- GEMM6: x2 @ W_obs_stat (K=512, N=1024) -> posterior head
        gemm_rows16<4, false>(s_v512, 512, 512, nullptr, 0,
                              wt_obs_stat, 512, s_acc, 1024, wave, lane);
        __syncthreads();
        stat_head(s_acc + w * 1024, b_obs_stat, outp + 0, outp + 1024,
                  s_idx + w * 32, lane);
        __syncthreads();
    }
}

// ---------------------------------------------------------------------------
extern "C" void kernel_launch(void* const* d_in, const int* in_sizes, int n_in,
                              void* d_out, int out_size, void* d_ws, size_t ws_size,
                              hipStream_t stream) {
    const float* embed     = (const float*)d_in[0];
    const float* action    = (const float*)d_in[1];
    const float* is_first  = (const float*)d_in[2];
    const float* W_img_in  = (const float*)d_in[3];
    const float* g_img_in  = (const float*)d_in[4];
    const float* b_img_in  = (const float*)d_in[5];
    const float* W_gru     = (const float*)d_in[6];
    const float* g_gru     = (const float*)d_in[7];
    const float* b_gru     = (const float*)d_in[8];
    const float* W_img_out = (const float*)d_in[9];
    const float* g_img_out = (const float*)d_in[10];
    const float* b_img_out = (const float*)d_in[11];
    const float* W_obs_out = (const float*)d_in[12];
    const float* g_obs_out = (const float*)d_in[13];
    const float* b_obs_out = (const float*)d_in[14];
    const float* W_ims_stat= (const float*)d_in[15];
    const float* b_ims_stat= (const float*)d_in[16];
    const float* W_obs_stat= (const float*)d_in[17];
    const float* b_obs_stat= (const float*)d_in[18];
    const float* W_init    = (const float*)d_in[19];

    char* ws = (char*)d_ws;
    size_t off = 0;
    auto take = [&](size_t bytes) -> void* {
        void* p = ws + off;
        off = (off + bytes + 255) & ~(size_t)255;
        return p;
    };
    bf16* wt_gru      = (bf16*)take((size_t)1536 * 1024 * 2);
    bf16* wt_img_out  = (bf16*)take((size_t)512  * 512  * 2);
    bf16* wt_ims_stat = (bf16*)take((size_t)1024 * 512  * 2);
    bf16* wt_obs_out  = (bf16*)take((size_t)512  * 1536 * 2);
    bf16* wt_obs_stat = (bf16*)take((size_t)1024 * 512  * 2);
    float* deter0     = (float*)take(512 * 4);
    int*   idx0       = (int*)take(32 * 4);

    // weight repack: f32 [K][N] -> bf16 [N][K]
    transpose_bf16<<<(1024 * 1536 + 255) / 256, 256, 0, stream>>>(W_gru,      wt_gru,      1024, 1536);
    transpose_bf16<<<(512  * 512  + 255) / 256, 256, 0, stream>>>(W_img_out,  wt_img_out,  512,  512);
    transpose_bf16<<<(512  * 1024 + 255) / 256, 256, 0, stream>>>(W_ims_stat, wt_ims_stat, 512,  1024);
    transpose_bf16<<<(1536 * 512  + 255) / 256, 256, 0, stream>>>(W_obs_out,  wt_obs_out,  1536, 512);
    transpose_bf16<<<(512  * 1024 + 255) / 256, 256, 0, stream>>>(W_obs_stat, wt_obs_stat, 512,  1024);

    init_state<<<1, 512, 0, stream>>>(W_init, W_img_out, g_img_out, b_img_out,
                                      W_ims_stat, b_ims_stat, deter0, idx0);

    rssm_main<<<dim3(32), dim3(512), 0, stream>>>(
        embed, action, is_first, W_img_in,
        g_img_in, b_img_in, g_gru, b_gru, g_img_out, b_img_out,
        g_obs_out, b_obs_out, b_ims_stat, b_obs_stat,
        wt_gru, wt_img_out, wt_ims_stat, wt_obs_out, wt_obs_stat,
        deter0, idx0, (float*)d_out);
}
